// MultiheadAttention_10299331576136
// MI455X (gfx1250) — matrix operand order repaired and checked by reading, not compiled, not run on hardware
//
#include <hip/hip_runtime.h>
#include <hip/hip_bf16.h>

// Problem constants (match reference)
#define BB 2
#define SS 2048
#define DD 1024
#define HH 16
#define HD 64

typedef __attribute__((ext_vector_type(16))) __bf16 v16bf;
typedef __attribute__((ext_vector_type(8)))  __bf16 v8bf;
typedef __attribute__((ext_vector_type(4)))  __bf16 v4bf;
typedef __attribute__((ext_vector_type(8)))  float  v8f;
typedef __attribute__((ext_vector_type(4)))  float  v4f;

// ---------------------------------------------------------------------------
// Async global->LDS staging (gfx1250 GLOBAL_LOAD_ASYNC_TO_LDS_B128, ASYNCcnt)
// Builtin signature (from hipcc diagnostic): param0 = AS(1) int4*, so:
//   (int4 AS1* gsrc, int4 AS3* ldst, imm offset, imm cpol)
// Guarded: falls back to VGPR round-trip if the builtins aren't declared.
// ---------------------------------------------------------------------------
#if defined(__has_builtin)
#  if __has_builtin(__builtin_amdgcn_global_load_async_to_lds_b128) && \
      __has_builtin(__builtin_amdgcn_s_wait_asynccnt)
#    define HAVE_ASYNC_LDS 1
#  endif
#endif
#ifndef HAVE_ASYNC_LDS
#  define HAVE_ASYNC_LDS 0
#endif

#if HAVE_ASYNC_LDS
typedef int v4i_vs __attribute__((vector_size(4 * sizeof(int))));
typedef __attribute__((address_space(1))) v4i_vs* gp_v4i;   // global int4*
typedef __attribute__((address_space(3))) v4i_vs* lp_v4i;   // LDS int4*
#endif

// copy 16 bytes (8 bf16) global -> LDS
__device__ __forceinline__ void copy16_g2l(const __bf16* g, __bf16* l) {
#if HAVE_ASYNC_LDS
  __builtin_amdgcn_global_load_async_to_lds_b128((gp_v4i)g, (lp_v4i)l, 0, 0);
#else
  *(v8bf*)l = *(const v8bf*)g;
#endif
}
__device__ __forceinline__ void async_join() {
#if HAVE_ASYNC_LDS
  __builtin_amdgcn_s_wait_asynccnt(0);
#endif
}

__device__ __forceinline__ v16bf pack16(v8bf lo, v8bf hi) {
  v16bf r;
#pragma unroll
  for (int i = 0; i < 8; ++i) { r[i] = lo[i]; r[i + 8] = hi[i]; }
  return r;
}

__device__ __forceinline__ v8f vzero8() {
  v8f z;
#pragma unroll
  for (int i = 0; i < 8; ++i) z[i] = 0.0f;
  return z;
}

// reduce across the 16 lanes holding one C-fragment row (wave32, width 16)
__device__ __forceinline__ float rmax16(float v) {
  v = fmaxf(v, __shfl_xor(v, 1, 16));
  v = fmaxf(v, __shfl_xor(v, 2, 16));
  v = fmaxf(v, __shfl_xor(v, 4, 16));
  v = fmaxf(v, __shfl_xor(v, 8, 16));
  return v;
}
__device__ __forceinline__ float rsum16(float v) {
  v += __shfl_xor(v, 1, 16);
  v += __shfl_xor(v, 2, 16);
  v += __shfl_xor(v, 4, 16);
  v += __shfl_xor(v, 8, 16);
  return v;
}

// ---------------------------------------------------------------------------
// fp32 -> bf16 cast (vectorized b128 load / b64 store)
// ---------------------------------------------------------------------------
__global__ __launch_bounds__(256) void cast_f32_bf16(const float* __restrict__ src,
                                                     __bf16* __restrict__ dst, int n) {
  int i = (blockIdx.x * 256 + threadIdx.x) * 4;
  if (i < n) {
    v4f f = *(const v4f*)(src + i);
    v4bf o;
#pragma unroll
    for (int j = 0; j < 4; ++j) o[j] = (__bf16)f[j];
    *(v4bf*)(dst + i) = o;
  }
}

// ---------------------------------------------------------------------------
// C[M x 1024] = A[M x 1024] @ W^T, W is [1024 x 1024] row-major (B[k][n]=W[n][k])
// 256 threads = 8 waves; block tile 64M x 64N; wave tile 16M x 32N.
// K staged through LDS in 32-wide panels (async copy); bf16 WMMA, f32 accum.
// ---------------------------------------------------------------------------
template <bool OUT_BF16>
__global__ __launch_bounds__(256) void gemm_xwT(const __bf16* __restrict__ A,
                                                const __bf16* __restrict__ W,
                                                __bf16* __restrict__ outB,
                                                float* __restrict__ outF) {
  const int K = DD, N = DD;
  __shared__ __align__(16) __bf16 As[64 * 32];   // [m][k]
  __shared__ __align__(16) __bf16 Bs[64 * 32];   // [n][k]

  const int m0 = blockIdx.y * 64;
  const int n0 = blockIdx.x * 64;
  const int t = threadIdx.x;
  const int w = t >> 5, lane = t & 31;
  const int lm = lane & 15, half = lane >> 4;
  const int wm = w >> 1, wn = w & 1;           // 4x2 wave grid

  v8f acc[2] = {vzero8(), vzero8()};

  const int srow = t >> 2;                     // 0..63
  const int sc8 = (t & 3) * 8;                 // 0,8,16,24

  for (int kk = 0; kk < K; kk += 32) {
    // cooperative stage of 64x32 A panel and 64x32 W panel (both k-contiguous)
    copy16_g2l(A + (size_t)(m0 + srow) * K + kk + sc8, As + srow * 32 + sc8);
    copy16_g2l(W + (size_t)(n0 + srow) * K + kk + sc8, Bs + srow * 32 + sc8);
    if (kk + 32 < K) {
      __builtin_prefetch(A + (size_t)(m0 + srow) * K + kk + 32 + sc8, 0, 0);
      __builtin_prefetch(W + (size_t)(n0 + srow) * K + kk + 32 + sc8, 0, 0);
    }
    async_join();
    __syncthreads();

    // A fragment: lane(l<16): M=lm, K elems {8h..8h+7, 16+8h..16+8h+7}
    const __bf16* ap = As + (16 * wm + lm) * 32;
    v16bf afrag = pack16(*(const v8bf*)(ap + 8 * half),
                         *(const v8bf*)(ap + 16 + 8 * half));
#pragma unroll
    for (int nt = 0; nt < 2; ++nt) {
      // B fragment: N=lm(+16nt), K elems {8h..8h+7, 16+8h..16+8h+7}
      const __bf16* bp = Bs + (32 * wn + 16 * nt + lm) * 32 + 8 * half;
      v16bf bfrag = pack16(*(const v8bf*)(bp), *(const v8bf*)(bp + 16));
      acc[nt] = __builtin_amdgcn_wmma_f32_16x16x32_bf16(
          false, afrag, false, bfrag, (short)0, acc[nt], false, false);
    }
    __syncthreads();
  }

  // Epilogue: C layout — VGPR r -> M = r + 8*half, N = lm
#pragma unroll
  for (int nt = 0; nt < 2; ++nt) {
#pragma unroll
    for (int r = 0; r < 8; ++r) {
      size_t row = (size_t)(m0 + 16 * wm + r + 8 * half);
      size_t col = (size_t)(n0 + 32 * wn + 16 * nt + lm);
      float v = acc[nt][r];
      if (OUT_BF16) outB[row * N + col] = (__bf16)v;
      else          outF[row * N + col] = v;
    }
  }
}

// ---------------------------------------------------------------------------
// Flash attention: block = (b, h, 128 query rows); 8 waves x 16-row Q tiles.
// KV processed in 32-row steps: scores via 2x2 WMMA, online softmax,
// P restaged through LDS (C-layout f32 -> A-layout bf16), O += P@V (4 WMMA).
// K panel staged with async copy; V panel transposed through registers.
// ---------------------------------------------------------------------------
__global__ __launch_bounds__(256) void attn_flash(const __bf16* __restrict__ qP,
                                                  const __bf16* __restrict__ kP,
                                                  const __bf16* __restrict__ vP,
                                                  const float* __restrict__ gamma,
                                                  __bf16* __restrict__ attn_out) {
  __shared__ __align__(16) __bf16 Ks[32 * 64];       // [j][c]
  __shared__ __align__(16) __bf16 Vt[64 * 32];       // [c][j] (transposed)
  __shared__ __align__(16) __bf16 Ps[8 * 16 * 32];   // per-wave P staging

  const int bid = blockIdx.x;
  const int qb = bid & 15;            // S/128 = 16
  const int h  = (bid >> 4) & 15;
  const int b  = bid >> 8;
  const int t = threadIdx.x;
  const int w = t >> 5, lane = t & 31;
  const int lm = lane & 15, half = lane >> 4;
  const int q0 = qb * 128 + w * 16;

  const float scale_qk = 0.125f;      // 1/sqrt(64)

  // Q fragments (persist in registers): two K-steps of 32 over HD=64
  v16bf aq[2];
  {
    const __bf16* qrow = qP + ((size_t)(b * SS + q0 + lm) * DD) + h * HD;
#pragma unroll
    for (int ks = 0; ks < 2; ++ks)
      aq[ks] = pack16(*(const v8bf*)(qrow + 32 * ks + 8 * half),
                      *(const v8bf*)(qrow + 32 * ks + 16 + 8 * half));
  }

  v8f o[4] = {vzero8(), vzero8(), vzero8(), vzero8()};
  float mrow[8], lrow[8];
#pragma unroll
  for (int r = 0; r < 8; ++r) { mrow[r] = -INFINITY; lrow[r] = 0.0f; }

  const int sj = t >> 3;              // 0..31
  const int sc8 = (t & 7) * 8;        // 0..56

  for (int kv0 = 0; kv0 < SS; kv0 += 32) {
    // stage K panel [32 x 64] row-major (async), V panel transposed [64 x 32]
    {
      size_t base = ((size_t)(b * SS + kv0 + sj) * DD) + h * HD + sc8;
      copy16_g2l(kP + base, Ks + sj * 64 + sc8);
      v8bf vr = *(const v8bf*)(vP + base);
#pragma unroll
      for (int i = 0; i < 8; ++i) Vt[(sc8 + i) * 32 + sj] = vr[i];
    }
    async_join();
    __syncthreads();

    // scores S[16 x 32] = Q[16 x 64] @ K^T
    v8f s[2] = {vzero8(), vzero8()};
#pragma unroll
    for (int ks = 0; ks < 2; ++ks) {
#pragma unroll
      for (int nt = 0; nt < 2; ++nt) {
        const __bf16* bp = Ks + (lm + 16 * nt) * 64 + 32 * ks + 8 * half;
        v16bf bk = pack16(*(const v8bf*)(bp), *(const v8bf*)(bp + 16));
        s[nt] = __builtin_amdgcn_wmma_f32_16x16x32_bf16(
            false, aq[ks], false, bk, (short)0, s[nt], false, false);
      }
    }

    // online softmax + write P tile (bf16) into per-wave LDS staging
    __bf16* pw = Ps + w * (16 * 32);
#pragma unroll
    for (int r = 0; r < 8; ++r) {
      float v0 = s[0][r] * scale_qk;
      float v1 = s[1][r] * scale_qk;
      float tmax = rmax16(fmaxf(v0, v1));
      float mnew = fmaxf(mrow[r], tmax);
      float corr = __expf(mrow[r] - mnew);
      float p0 = __expf(v0 - mnew);
      float p1 = __expf(v1 - mnew);
      lrow[r] = lrow[r] * corr + rsum16(p0 + p1);
      mrow[r] = mnew;
      pw[(r + 8 * half) * 32 + lm]      = (__bf16)p0;
      pw[(r + 8 * half) * 32 + 16 + lm] = (__bf16)p1;
#pragma unroll
      for (int nt2 = 0; nt2 < 4; ++nt2) o[nt2][r] *= corr;
    }
    __syncthreads();

    // O[16 x 64] += P[16 x 32] @ V[32 x 64]
    const __bf16* pp = pw + lm * 32;
    v16bf ap = pack16(*(const v8bf*)(pp + 8 * half),
                      *(const v8bf*)(pp + 16 + 8 * half));
#pragma unroll
    for (int nt2 = 0; nt2 < 4; ++nt2) {
      const __bf16* vpp = Vt + (lm + 16 * nt2) * 32 + 8 * half;
      v16bf bv = pack16(*(const v8bf*)(vpp), *(const v8bf*)(vpp + 16));
      o[nt2] = __builtin_amdgcn_wmma_f32_16x16x32_bf16(
          false, ap, false, bv, (short)0, o[nt2], false, false);
    }
    __syncthreads();
  }

  // epilogue: 1/l normalization, per-head gamma, merged-head [B,S,D] bf16 store
  const float g = gamma[h];
#pragma unroll
  for (int nt2 = 0; nt2 < 4; ++nt2) {
#pragma unroll
    for (int r = 0; r < 8; ++r) {
      float val = o[nt2][r] / lrow[r] * g;
      size_t row = (size_t)(b * SS + q0 + r + 8 * half);
      attn_out[row * DD + h * HD + 16 * nt2 + lm] = (__bf16)val;
    }
  }
}

// ---------------------------------------------------------------------------
// Launcher
// ---------------------------------------------------------------------------
extern "C" void kernel_launch(void* const* d_in, const int* in_sizes, int n_in,
                              void* d_out, int out_size, void* d_ws, size_t ws_size,
                              hipStream_t stream) {
  const float* query = (const float*)d_in[0];
  const float* key   = (const float*)d_in[1];
  const float* value = (const float*)d_in[2];
  const float* Wq    = (const float*)d_in[3];
  const float* Wk    = (const float*)d_in[4];
  const float* Wv    = (const float*)d_in[5];
  const float* Wo    = (const float*)d_in[6];
  const float* gamma = (const float*)d_in[7];

  const int NX = BB * SS * DD;   // 4,194,304
  const int NW = DD * DD;        // 1,048,576
  const size_t XBYT = (size_t)NX * sizeof(__bf16);
  const size_t WBYT = (size_t)NW * sizeof(__bf16);

  char* ws = (char*)d_ws;
  size_t off = 0;
  auto carve = [&](size_t bytes) { char* p = ws + off; off += bytes; return p; };
  __bf16* xq  = (__bf16*)carve(XBYT);
  __bf16* xk  = (__bf16*)carve(XBYT);
  __bf16* xv  = (__bf16*)carve(XBYT);
  __bf16* wqb = (__bf16*)carve(WBYT);
  __bf16* wkb = (__bf16*)carve(WBYT);
  __bf16* wvb = (__bf16*)carve(WBYT);
  __bf16* wob = (__bf16*)carve(WBYT);
  __bf16* qp  = (__bf16*)carve(XBYT);
  __bf16* kp  = (__bf16*)carve(XBYT);
  __bf16* vp  = (__bf16*)carve(XBYT);
  __bf16* ao  = (__bf16*)carve(XBYT);

  // 1) casts
  cast_f32_bf16<<<NX / 1024, 256, 0, stream>>>(query, xq, NX);
  cast_f32_bf16<<<NX / 1024, 256, 0, stream>>>(key,   xk, NX);
  cast_f32_bf16<<<NX / 1024, 256, 0, stream>>>(value, xv, NX);
  cast_f32_bf16<<<NW / 1024, 256, 0, stream>>>(Wq, wqb, NW);
  cast_f32_bf16<<<NW / 1024, 256, 0, stream>>>(Wk, wkb, NW);
  cast_f32_bf16<<<NW / 1024, 256, 0, stream>>>(Wv, wvb, NW);
  cast_f32_bf16<<<NW / 1024, 256, 0, stream>>>(Wo, wob, NW);

  // 2) projections: q/k/v = x @ W^T  (bf16 out)
  dim3 ggrid(DD / 64, (BB * SS) / 64);   // 16 x 64
  gemm_xwT<true><<<ggrid, 256, 0, stream>>>(xq, wqb, qp, nullptr);
  gemm_xwT<true><<<ggrid, 256, 0, stream>>>(xk, wkb, kp, nullptr);
  gemm_xwT<true><<<ggrid, 256, 0, stream>>>(xv, wvb, vp, nullptr);

  // 3) flash attention (B*H*(S/128) = 512 blocks)
  attn_flash<<<BB * HH * (SS / 128), 256, 0, stream>>>(qp, kp, vp, gamma, ao);

  // 4) output projection: out = attn @ Wo^T  (fp32 out)
  gemm_xwT<false><<<ggrid, 256, 0, stream>>>(ao, wob, nullptr, (float*)d_out);
}
